// Attention_77704548319841
// MI455X (gfx1250) — matrix operand order repaired and checked by reading, not compiled, not run on hardware
//
#include <hip/hip_runtime.h>
#include <stdint.h>

// ---------------------------------------------------------------------------
// CDNA5 (gfx1250) bf16-WMMA attention pipeline, round 3 (prefetch-scope fix).
// B=16, T=1024, S=1024, H=512, C2=2H=1024.
//
// Pipeline:
//   0) cvt f32->bf16: dec_bf, ctx_bf, wa_bf    (one-time; keeps GEMM inner
//      loops free of conversion VALU, halves panel traffic)
//   1) transpose ctx_bf -> ctxT [b][c][s]      (GEMM3 B operand, [N][K] layout)
//   2) GEMM1: ctxin = ctx @ Wa^T            (M=S,N=H,K=2H)  bf16 out
//   3) GEMM2: scores = dec @ ctxin^T + mask (M=T,N=S,K=H)   f32 out
//   4) softmax rows -> attn bf16
//   5) GEMM3: out = attn @ ctx              (M=T,N=2H,K=S)  f32 out
//
// GEMM: 128x64 block tile, 8 wave32, wave tile 32x32 (4 WMMA accumulators),
// K-step 32, double-buffered LDS with register prefetch -> 1 barrier / K-step,
// WGP-scope global_prefetch for the K+2 tile.
// ---------------------------------------------------------------------------

typedef __attribute__((ext_vector_type(16))) __bf16 v16bf;
typedef __attribute__((ext_vector_type(8)))  float  v8f;

struct __align__(16) Q16 { unsigned int x, y, z, w; };   // POD 16-byte chunk
struct __align__(16) F4  { float x, y, z, w; };          // POD float4

union FragU { v16bf v; Q16 q[2]; };

static __device__ __forceinline__ unsigned short f32_to_bf16_rne(float f) {
    union { float f; unsigned int u; } x; x.f = f;
    unsigned int r = x.u + 0x7FFFu + ((x.u >> 16) & 1u);
    return (unsigned short)(r >> 16);
}

static __device__ __forceinline__ unsigned int pack2_bf16(float lo, float hi) {
    return (unsigned int)f32_to_bf16_rne(lo) | ((unsigned int)f32_to_bf16_rne(hi) << 16);
}

// --- Fragment loads from LDS (layouts per CDNA5 ISA 7.12.2) ---------------
// A: 16x32 bf16. lane L: row = L%16; K-chunks [kb,kb+8) and [kb+16,kb+24), kb=8*(L>=16).
static __device__ __forceinline__ v16bf load_a_frag(const unsigned short* tile,
                                                    int row_base, int lane) {
    int r  = row_base + (lane & 15);
    int kb = (lane >> 4) * 8;
    FragU f;
    f.q[0] = *(const Q16*)(tile + r * 32 + kb);
    f.q[1] = *(const Q16*)(tile + r * 32 + kb + 16);
    return f.v;
}

// B (stored as B^T row-major [N][K]): lane L: col n = L%16; K-chunks [kb,kb+8) and [kb+16,kb+24), kb=8*(L>=16).
static __device__ __forceinline__ v16bf load_b_frag(const unsigned short* tile,
                                                    int col_base, int lane) {
    int n  = col_base + (lane & 15);
    int kb = (lane >> 4) * 8;
    FragU f;
    f.q[0] = *(const Q16*)(tile + n * 32 + kb);
    f.q[1] = *(const Q16*)(tile + n * 32 + kb + 16);
    return f.v;
}

#define WMMA_BF16(a, b, c) \
    __builtin_amdgcn_wmma_f32_16x16x32_bf16(false, (a), false, (b), (short)0, (c), false, false)

// ---------------------------------------------------------------------------
// Batched bf16 WMMA GEMM:  C[b] (MxN) = A[b] (MxK, row-major) * Bt[b] (NxK)^T
// EPI: 0 = store bf16, 1 = store f32 with mask (-1e6 where mask!=0), 2 = store f32
// ---------------------------------------------------------------------------
template <int EPI>
__global__ __launch_bounds__(256)
void wmma_gemm(const unsigned short* __restrict__ A,  long long lda, long long strideA,
               const unsigned short* __restrict__ Bt, long long ldb, long long strideB,
               void* __restrict__ C,                  long long ldc, long long strideC,
               const unsigned char* __restrict__ mask, int K)
{
    __shared__ unsigned short As[2][128 * 32];
    __shared__ unsigned short Bs[2][64 * 32];

    const int b    = blockIdx.z;
    const int tid  = threadIdx.x;
    const int wave = tid >> 5;
    const int lane = tid & 31;

    const int m0 = blockIdx.x * 128;
    const int n0 = blockIdx.y * 64;
    const int wm = (wave & 3) * 32;   // wave 32-row band within block tile
    const int wn = (wave >> 2) * 32;  // wave 32-col band within block tile

    // Staging coordinates: A tile 128x32 (2 x Q16 per thread), B tile 64x32 (1 x Q16).
    const int ar = tid >> 1;          // 0..127
    const int ac = (tid & 1) * 16;    // 0 / 16
    const int br = tid >> 2;          // 0..63
    const int bc = (tid & 3) * 8;     // 0,8,16,24

    const unsigned short* ag = A  + strideA * b + (long long)m0 * lda + (long long)ar * lda + ac;
    const unsigned short* bg = Bt + strideB * b + (long long)n0 * ldb + (long long)br * ldb + bc;

    // Prime the pipeline: stage K-tile 0 into buffer 0.
    Q16 pa0 = *(const Q16*)(ag);
    Q16 pa1 = *(const Q16*)(ag + 8);
    Q16 pb  = *(const Q16*)(bg);
    *(Q16*)(&As[0][ar * 32 + ac])     = pa0;
    *(Q16*)(&As[0][ar * 32 + ac + 8]) = pa1;
    *(Q16*)(&Bs[0][br * 32 + bc])     = pb;
    __syncthreads();

    v8f acc00 = {}, acc01 = {}, acc10 = {}, acc11 = {};

    int buf = 0;
    for (int k0 = 0; k0 < K; k0 += 32, buf ^= 1) {
        const int  kn   = k0 + 32;
        const bool more = kn < K;

        // Register prefetch of the next K-tile (overlaps with WMMA below).
        if (more) {
            pa0 = *(const Q16*)(ag + kn);
            pa1 = *(const Q16*)(ag + kn + 8);
            pb  = *(const Q16*)(bg + kn);
        }
        // Deep prefetch of the K+2 tile, WGP scope (locality 3 -> near caches).
        if (kn + 32 < K) {
            __builtin_prefetch((const void*)(ag + kn + 32), 0, 3);
            __builtin_prefetch((const void*)(bg + kn + 32), 0, 3);
        }

        const unsigned short* Ac = As[buf];
        const unsigned short* Bc = Bs[buf];
        v16bf af0 = load_a_frag(Ac, wm,      lane);
        v16bf af1 = load_a_frag(Ac, wm + 16, lane);
        v16bf bf0 = load_b_frag(Bc, wn,      lane);
        v16bf bf1 = load_b_frag(Bc, wn + 16, lane);

        acc00 = WMMA_BF16(af0, bf0, acc00);
        acc01 = WMMA_BF16(af0, bf1, acc01);
        acc10 = WMMA_BF16(af1, bf0, acc10);
        acc11 = WMMA_BF16(af1, bf1, acc11);

        // Store the prefetched tile into the other buffer, then one barrier.
        if (more) {
            unsigned short* An = As[buf ^ 1];
            unsigned short* Bn = Bs[buf ^ 1];
            *(Q16*)(&An[ar * 32 + ac])     = pa0;
            *(Q16*)(&An[ar * 32 + ac + 8]) = pa1;
            *(Q16*)(&Bn[br * 32 + bc])     = pb;
        }
        __syncthreads();
    }

    // C/D layout: VGPR r, lane L -> row r + 8*(L>=16), col L%16.
    const int n       = lane & 15;
    const int row_off = (lane >> 4) * 8;

    v8f accs[2][2] = { { acc00, acc01 }, { acc10, acc11 } };
    for (int mi = 0; mi < 2; ++mi) {
        for (int ni = 0; ni < 2; ++ni) {
            for (int r = 0; r < 8; ++r) {
                const long long gm = m0 + wm + 16 * mi + r + row_off;
                const long long gn = n0 + wn + 16 * ni + n;
                const float     v  = accs[mi][ni][r];
                if (EPI == 0) {
                    unsigned short* Cp = (unsigned short*)C + strideC * b;
                    Cp[gm * ldc + gn] = f32_to_bf16_rne(v);
                } else if (EPI == 1) {
                    float* Cp = (float*)C + strideC * b;
                    const unsigned char* Mp = mask + strideC * b;  // same [T][S] geometry
                    Cp[gm * ldc + gn] = Mp[gm * ldc + gn] ? -1000000.0f : v;
                } else {
                    float* Cp = (float*)C + strideC * b;
                    Cp[gm * ldc + gn] = v;
                }
            }
        }
    }
}

// ---------------------------------------------------------------------------
// Elementwise f32 -> bf16 (n divisible by 2048; 8 elements / thread).
// ---------------------------------------------------------------------------
__global__ __launch_bounds__(256)
void cvt_f32_bf16_kernel(const float* __restrict__ src, unsigned short* __restrict__ dst)
{
    const long long i = ((long long)blockIdx.x * 256 + threadIdx.x) * 8;
    F4 f0 = *(const F4*)(src + i);
    F4 f1 = *(const F4*)(src + i + 4);
    Q16 o;
    o.x = pack2_bf16(f0.x, f0.y);
    o.y = pack2_bf16(f0.z, f0.w);
    o.z = pack2_bf16(f1.x, f1.y);
    o.w = pack2_bf16(f1.z, f1.w);
    *(Q16*)(dst + i) = o;
}

// ---------------------------------------------------------------------------
// Transpose bf16: ctx_bf[b][s][c] -> ctxT[b][c][s]
// ---------------------------------------------------------------------------
__global__ __launch_bounds__(256)
void transpose_kernel(const unsigned short* __restrict__ src_all,
                      unsigned short* __restrict__ dst_all, int S, int C)
{
    __shared__ unsigned short tile[32][33];
    const int b = blockIdx.z;
    const unsigned short* src = src_all + (long long)b * S * C;
    unsigned short*       dst = dst_all + (long long)b * C * S;
    const int s0 = blockIdx.x * 32;
    const int c0 = blockIdx.y * 32;
    const int tx = threadIdx.x & 31;
    const int ty = threadIdx.x >> 5;   // 0..7
    for (int i = 0; i < 4; ++i)
        tile[ty + i * 8][tx] = src[(long long)(s0 + ty + i * 8) * C + c0 + tx];
    __syncthreads();
    for (int i = 0; i < 4; ++i)
        dst[(long long)(c0 + ty + i * 8) * S + s0 + tx] = tile[tx][ty + i * 8];
}

// ---------------------------------------------------------------------------
// Row softmax: one 256-thread block per (b,t) row of length S. f32 in, bf16 out.
// ---------------------------------------------------------------------------
__global__ __launch_bounds__(256)
void softmax_kernel(const float* __restrict__ scores, unsigned short* __restrict__ attn, int S)
{
    __shared__ float red[256];
    const long long row = blockIdx.x;
    const float* src = scores + row * S;
    unsigned short* dst = attn + row * S;
    const int tid = threadIdx.x;

    float m = -3.4e38f;
    for (int s = tid; s < S; s += 256) m = fmaxf(m, src[s]);
    red[tid] = m;
    __syncthreads();
    for (int st = 128; st > 0; st >>= 1) {
        if (tid < st) red[tid] = fmaxf(red[tid], red[tid + st]);
        __syncthreads();
    }
    m = red[0];
    __syncthreads();

    float sum = 0.0f;
    for (int s = tid; s < S; s += 256) sum += __expf(src[s] - m);
    red[tid] = sum;
    __syncthreads();
    for (int st = 128; st > 0; st >>= 1) {
        if (tid < st) red[tid] = red[tid] + red[tid + st];
        __syncthreads();
    }
    const float inv = 1.0f / red[0];
    __syncthreads();

    for (int s = tid; s < S; s += 256)
        dst[s] = f32_to_bf16_rne(__expf(src[s] - m) * inv);
}

// ---------------------------------------------------------------------------
extern "C" void kernel_launch(void* const* d_in, const int* in_sizes, int n_in,
                              void* d_out, int out_size, void* d_ws, size_t ws_size,
                              hipStream_t stream)
{
    (void)in_sizes; (void)n_in; (void)out_size; (void)ws_size;

    constexpr int B = 16, T = 1024, S = 1024, H = 512, C2 = 1024;

    const float*         dec  = (const float*)d_in[0];          // [B][T][H]
    const float*         ctx  = (const float*)d_in[1];          // [B][S][2H]
    const unsigned char* mask = (const unsigned char*)d_in[2];  // [B][T][S] bool
    const float*         Wa   = (const float*)d_in[3];          // [H][2H]
    float*               out  = (float*)d_out;                  // [B][T][2H]

    // Workspace layout (all 16B-aligned):
    char* ws = (char*)d_ws;
    size_t off = 0;
    unsigned short* ctx_bf = (unsigned short*)(ws + off); off += (size_t)B * S * C2 * 2; // 33.5 MB
    unsigned short* dec_bf = (unsigned short*)(ws + off); off += (size_t)B * T * H  * 2; // 16.8 MB
    unsigned short* wa_bf  = (unsigned short*)(ws + off); off += (size_t)H * C2     * 2; //  1.0 MB
    unsigned short* ctxin  = (unsigned short*)(ws + off); off += (size_t)B * S * H  * 2; // 16.8 MB
    unsigned short* ctxT   = (unsigned short*)(ws + off); off += (size_t)B * C2 * S * 2; // 33.5 MB
    unsigned short* attn   = (unsigned short*)(ws + off); off += (size_t)B * T * S  * 2; // 33.5 MB
    float*          scor   = (float*)(ws + off);          off += (size_t)B * T * S  * 4; // 67.1 MB

    dim3 blk(256);

    // 0) One-time f32 -> bf16 conversions.
    cvt_f32_bf16_kernel<<<dim3(((size_t)B * S * C2) / 2048), blk, 0, stream>>>(ctx, ctx_bf);
    cvt_f32_bf16_kernel<<<dim3(((size_t)B * T * H)  / 2048), blk, 0, stream>>>(dec, dec_bf);
    cvt_f32_bf16_kernel<<<dim3(((size_t)H * C2)     / 2048), blk, 0, stream>>>(Wa,  wa_bf);

    // 1) ctxT[b][c][s] = ctx_bf[b][s][c]
    transpose_kernel<<<dim3(S / 32, C2 / 32, B), blk, 0, stream>>>(ctx_bf, ctxT, S, C2);

    // 2) ctxin[b][s][h] = sum_c ctx[b][s][c] * Wa[h][c]   (M=S, N=H, K=2H)
    wmma_gemm<0><<<dim3(S / 128, H / 64, B), blk, 0, stream>>>(
        ctx_bf, C2, (long long)S * C2,
        wa_bf,  C2, 0,
        ctxin,  H,  (long long)S * H,
        nullptr, C2);

    // 3) scores[b][t][s] = sum_h dec[b][t][h] * ctxin[b][s][h], masked  (M=T, N=S, K=H)
    wmma_gemm<1><<<dim3(T / 128, S / 64, B), blk, 0, stream>>>(
        dec_bf, H, (long long)T * H,
        ctxin,  H, (long long)S * H,
        scor,   S, (long long)T * S,
        mask, H);

    // 4) attn = softmax(scores, axis=S), bf16
    softmax_kernel<<<dim3(B * T), blk, 0, stream>>>(scor, attn, S);

    // 5) out[b][t][c] = sum_s attn[b][t][s] * ctx[b][s][c]   (M=T, N=2H, K=S)
    wmma_gemm<2><<<dim3(T / 128, C2 / 64, B), blk, 0, stream>>>(
        attn, S, (long long)T * S,
        ctxT, S, (long long)C2 * S,
        out,  C2, (long long)T * C2,
        nullptr, S);
}
